// AttentionGrouping_37297495998975
// MI455X (gfx1250) — matrix operand order repaired and checked by reading, not compiled, not run on hardware
//
#include <hip/hip_runtime.h>
#include <hip/hip_bf16.h>

// ---------------------------------------------------------------------------
// Fused sparsemax graph-attention for MI455X (gfx1250, wave32, WMMA).
//   inputs : node_feature [4096,32,128] f32, WK, WV, WQ [128,256] f32
//   outputs: output [131072,128] f32 ++ weight [131072,32,2] f32 (concat)
// Strategy: bf16 WMMA (f32 accumulate) for all three GEMM stages -> memory
// bound (~170MB @ 23.3TB/s); fp32 WMMA would be ~7x compute bound.
// One workgroup (256 thr = 8 waves) per graph; everything fused in LDS.
// d_ws usage: 192KB of pre-swizzled bf16 weights (B-operand layout).
// Wave-derived control values go through readfirstlane so tile selection is
// SALU/s_cbranch (EXEC stays all-ones around every WMMA, per ISA 7.12).
// ---------------------------------------------------------------------------

typedef unsigned int   u32;
typedef unsigned short u16;

typedef __bf16 v16bf __attribute__((ext_vector_type(16)));
typedef float  v8f   __attribute__((ext_vector_type(8)));

#define NGRAPH 4096
#define NNODE  32
#define EMB    128
#define NHEAD  2
#define HD     256          // NHEAD*EMB

// LDS geometry (ushort counts unless noted). Row strides chosen so every
// b128 LDS access is 16B aligned and lane->bank mapping is conflict-free.
#define XBF_STRIDE 136      // 128 bf16 + pad   (row = 272B)
#define QK_STRIDE  264      // 256 bf16 + pad   (row = 528B)
#define VT_STRIDE  40       // 32 bf16 + pad    (row = 80B)
#define WB_STRIDE  40
#define SC_STRIDE  36       // fp32 scores row pad

#define XBF_N   (NNODE * XBF_STRIDE)            // 4352
#define Q_N     (NNODE * QK_STRIDE)             // 8448
#define K_N     (NNODE * QK_STRIDE)             // 8448
#define VT_N    (NHEAD * EMB * VT_STRIDE)       // 10240
#define WB_N    (NHEAD * NNODE * WB_STRIDE)     // 2560
#define SC_N    (NHEAD * NNODE * SC_STRIDE)     // 2304 floats
#define SMEM_BYTES ((XBF_N + Q_N + K_N + VT_N + WB_N) * 2 + SC_N * 4)  // 77312

__device__ __forceinline__ u16 f2bf_bits(float f) {  // RNE f32 -> bf16
  u32 u = __float_as_uint(f);
  u += 0x7fffu + ((u >> 16) & 1u);
  return (u16)(u >> 16);
}

union V16BF_U { uint4 q[2]; v16bf v; };

// A operand, 16x32 bf16 (M x K) from row-major bf16 [rows][rowStride].
// lanes 0-15: M=lane, K=0..15 contiguous
//             lanes16-31: M=lane-16, K=16..31 contiguous
__device__ __forceinline__ v16bf load_tileA(const u16* p, int rowStride,
                                            int m0, int k0) {
  int lane = threadIdx.x & 31;
  const u16* row = p + (m0 + (lane & 15)) * rowStride + k0 + ((lane < 16) ? 0 : 16);
  V16BF_U u;
  u.q[0] = *(const uint4*)(row);        // K = kb .. kb+7
  u.q[1] = *(const uint4*)(row + 8);    // K = kb+8 .. kb+15
  return u.v;
}

// B operand, 32x16 bf16 (K x N) where source is row-major [N][K] (i.e. B^T).
// k order used here (same as load_tileA): lanes 0-15: N=lane, dw j -> K=2j,2j+1 ; lanes 16-31: N=lane-16, K=16+2j..
__device__ __forceinline__ v16bf load_tileB_T(const u16* p, int rowStride,
                                              int n0, int k0) {
  int lane = threadIdx.x & 31;
  const u16* row = p + (n0 + (lane & 15)) * rowStride + k0 + ((lane < 16) ? 0 : 16);
  V16BF_U u;
  u.q[0] = *(const uint4*)(row);        // 8 consecutive K values
  u.q[1] = *(const uint4*)(row + 8);
  return u.v;
}

// B operand from pre-swizzled workspace tile: 32 lanes x 8 contiguous dwords.
__device__ __forceinline__ v16bf load_tileB_ws(const u32* tile) {
  const u32* p = tile + (threadIdx.x & 31) * 8;
  V16BF_U u;
  u.q[0] = *(const uint4*)(p);
  u.q[1] = *(const uint4*)(p + 4);
  return u.v;
}

__device__ __forceinline__ v8f wmma_bf16(v16bf a, v16bf b, v8f c) {
  return __builtin_amdgcn_wmma_f32_16x16x32_bf16(false, a, false, b,
                                                 (short)0, c, false, false);
}

// C/D layout (ISA): VGPR j -> row r0 + j (+8 for lanes 16-31), col = c0 + lane%16.
__device__ __forceinline__ void store_C_bf16(v8f c, u16* p, int rowStride,
                                             int r0, int c0) {
  int lane = threadIdx.x & 31;
  int col = c0 + (lane & 15);
  int rb  = r0 + ((lane < 16) ? 0 : 8);
#pragma unroll
  for (int j = 0; j < 8; ++j) p[(rb + j) * rowStride + col] = f2bf_bits(c[j]);
}

__device__ __forceinline__ void store_C_bf16_T(v8f c, u16* p, int rowStride,
                                               int r0, int c0) {  // p[col][row]
  int lane = threadIdx.x & 31;
  u16* colp = p + (c0 + (lane & 15)) * rowStride + r0 + ((lane < 16) ? 0 : 8);
#pragma unroll
  for (int j = 0; j < 8; ++j) colp[j] = f2bf_bits(c[j]);
}

__device__ __forceinline__ void store_C_f32(v8f c, float* p, int rowStride,
                                            int r0, int c0, float scale) {
  int lane = threadIdx.x & 31;
  int col = c0 + (lane & 15);
  int rb  = r0 + ((lane < 16) ? 0 : 8);
#pragma unroll
  for (int j = 0; j < 8; ++j) p[(rb + j) * rowStride + col] = c[j] * scale;
}

// ---------------------------------------------------------------------------
// Prep: convert [WQ|WK|WV] f32 -> bf16 into d_ws, pre-swizzled as 192 tiles
// (48 n-tiles x 4 k-chunks) of 32x16 in exact B-operand per-lane order.
// ---------------------------------------------------------------------------
__global__ __launch_bounds__(256) void prep_weights(
    const float* __restrict__ WQ, const float* __restrict__ WK,
    const float* __restrict__ WV, u32* __restrict__ ws) {
  int wave = __builtin_amdgcn_readfirstlane(threadIdx.x >> 5);
  int lane = threadIdx.x & 31;
  int tid = blockIdx.x * 8 + wave;          // 0..191 (scalar)
  int nt = tid >> 2, kc = tid & 3;
  int mi = nt >> 4;                          // 0:Q 1:K 2:V (16 n-tiles each)
  const float* src = (mi == 0) ? WQ : (mi == 1) ? WK : WV;
  int n  = (nt & 15) * 16 + (lane & 15);     // column within [0,256)
  int kb = (lane < 16) ? 0 : 16;
  u32* dst = ws + tid * 256 + lane * 8;
#pragma unroll
  for (int j = 0; j < 8; ++j) {
    int k = kc * 32 + kb + 2 * j;
    u32 lo = f2bf_bits(src[k * HD + n]);
    u32 hi = f2bf_bits(src[(k + 1) * HD + n]);
    dst[j] = lo | (hi << 16);
  }
}

// ---------------------------------------------------------------------------
// Main fused kernel: one workgroup per graph.
// ---------------------------------------------------------------------------
__global__ __launch_bounds__(256) void attn_sparsemax_kernel(
    const float* __restrict__ x, const u32* __restrict__ wsw,
    float* __restrict__ outp, float* __restrict__ wout) {
  extern __shared__ char smem[];
  u16* xbf = (u16*)smem;                 // [32][136] bf16 node features
  u16* qld = xbf + XBF_N;                // [32][264] bf16 q (heads concat)
  u16* kld = qld + Q_N;                  // [32][264] bf16 k
  u16* vT  = kld + K_N;                  // [2][128][40] bf16 v transposed
  u16* wbf = vT + VT_N;                  // [2][32][40] bf16 sparsemax weights
  float* sc = (float*)(wbf + WB_N);      // [2][32][36] f32 scores

  const int b = blockIdx.x;
  const int t = threadIdx.x, lane = t & 31;
  // wave id is uniform across the wave32: scalarize all tile control flow.
  const int wave = __builtin_amdgcn_readfirstlane(t >> 5);

  // ---- load x[b] (32x128 f32), convert to bf16 rows in LDS ----
  const float4* xg = (const float4*)(x + (size_t)b * NNODE * EMB);
#pragma unroll
  for (int cc = 0; cc < 4; ++cc) {
    int c = t * 4 + cc;                  // float4 chunk id, 1024 total
    float4 f = xg[c];
    int fi = c * 4, row = fi >> 7, col = fi & 127;
    u32* p = (u32*)(xbf + row * XBF_STRIDE + col);
    p[0] = (u32)f2bf_bits(f.x) | ((u32)f2bf_bits(f.y) << 16);
    p[1] = (u32)f2bf_bits(f.z) | ((u32)f2bf_bits(f.w) << 16);
  }
  __syncthreads();

  // ---- stage 1: [q|k|v] = x @ [WQ|WK|WV]  (32 x 768, K=128) ----
  {
    const int mt = wave & 1, ncl = wave >> 1;  // 2 m-tiles x 4 n-clusters
    v16bf a0 = load_tileA(xbf, XBF_STRIDE, mt * 16, 0);
    v16bf a1 = load_tileA(xbf, XBF_STRIDE, mt * 16, 32);
    v16bf a2 = load_tileA(xbf, XBF_STRIDE, mt * 16, 64);
    v16bf a3 = load_tileA(xbf, XBF_STRIDE, mt * 16, 96);
#pragma unroll 2
    for (int it = 0; it < 12; ++it) {
      const int nt = ncl + it * 4;       // 0..47, scalar
      const u32* tb = wsw + (nt * 4) * 256;
      v8f acc = {0.f, 0.f, 0.f, 0.f, 0.f, 0.f, 0.f, 0.f};
      acc = wmma_bf16(a0, load_tileB_ws(tb), acc);
      acc = wmma_bf16(a1, load_tileB_ws(tb + 256), acc);
      acc = wmma_bf16(a2, load_tileB_ws(tb + 512), acc);
      acc = wmma_bf16(a3, load_tileB_ws(tb + 768), acc);
      if (nt < 16) {                     // q, row-major [node][256]
        store_C_bf16(acc, qld, QK_STRIDE, mt * 16, nt * 16);
      } else if (nt < 32) {              // k, row-major [node][256]
        store_C_bf16(acc, kld, QK_STRIDE, mt * 16, (nt - 16) * 16);
      } else {                           // v, transposed -> vT[h][d][node]
        int r = nt - 32, h = r >> 3, d0 = (r & 7) * 16;
        store_C_bf16_T(acc, vT + h * EMB * VT_STRIDE, VT_STRIDE, mt * 16, d0);
      }
    }
  }
  __syncthreads();

  // ---- stage 2: score[h] = q_h @ k_h^T / 16  (32x32, K=128) ----
  {
    const int h = wave >> 2, mt = (wave >> 1) & 1, ntj = wave & 1;  // scalar
    v8f acc = {0.f, 0.f, 0.f, 0.f, 0.f, 0.f, 0.f, 0.f};
#pragma unroll
    for (int kc = 0; kc < 4; ++kc) {
      v16bf a  = load_tileA  (qld, QK_STRIDE, mt * 16, h * EMB + kc * 32);
      v16bf bb = load_tileB_T(kld, QK_STRIDE, ntj * 16, h * EMB + kc * 32);
      acc = wmma_bf16(a, bb, acc);
    }
    store_C_f32(acc, sc + h * NNODE * SC_STRIDE, SC_STRIDE,
                mt * 16, ntj * 16, 0.0625f);  // 1/sqrt(D*H) = 1/16
  }
  __syncthreads();

  // ---- sparsemax over neighbor dim (exact, sort-free O(K^2)) ----
  if (wave < 2) {                        // scalar branch: waves 0-1 only
    int h = wave, i = lane;
    const float* z = sc + (h * NNODE + i) * SC_STRIDE;
    int ksup = 0; float ssup = 0.f;
    for (int e = 0; e < NNODE; ++e) {
      float ze = z[e]; int c = 0; float s = 0.f;
      for (int j = 0; j < NNODE; ++j) {
        float zj = z[j];
        if (zj >= ze) { ++c; s += zj; }
      }
      // support condition at (tie-run end of) sorted position c
      if (1.f + (float)c * ze > s && c > ksup) { ksup = c; ssup = s; }
    }
    float tau = (ssup - 1.f) / (float)ksup;
    float* wg = wout + ((size_t)(b * NNODE + i) * NNODE) * NHEAD + h;
    u16* wrow = wbf + (h * NNODE + i) * WB_STRIDE;
    for (int j = 0; j < NNODE; ++j) {
      float wv = z[j] - tau;
      wv = wv > 0.f ? wv : 0.f;
      wg[j * NHEAD] = wv;                // weight output [B*N, N, H]
      wrow[j] = f2bf_bits(wv);
    }
  }
  __syncthreads();

  // ---- stage 3: out = mean_h( w_h @ v_h )  (32x128, K=32) ----
  {
    float* g = outp + (size_t)b * NNODE * EMB;
#pragma unroll
    for (int t2 = 0; t2 < 2; ++t2) {
      const int tile = wave * 2 + t2;    // 16 (m,d) tiles / 8 waves, scalar
      const int mt = tile >> 3, dt = tile & 7;
      v8f acc = {0.f, 0.f, 0.f, 0.f, 0.f, 0.f, 0.f, 0.f};
#pragma unroll
      for (int h = 0; h < NHEAD; ++h) {
        v16bf a  = load_tileA  (wbf + h * NNODE * WB_STRIDE, WB_STRIDE, mt * 16, 0);
        v16bf bb = load_tileB_T(vT  + h * EMB   * VT_STRIDE, VT_STRIDE, dt * 16, 0);
        acc = wmma_bf16(a, bb, acc);
      }
      int col = dt * 16 + (lane & 15);
      int rb  = mt * 16 + ((lane < 16) ? 0 : 8);
#pragma unroll
      for (int j = 0; j < 8; ++j)
        g[(rb + j) * EMB + col] = acc[j] * 0.5f;   // mean over 2 heads
    }
  }
}

extern "C" void kernel_launch(void* const* d_in, const int* in_sizes, int n_in,
                              void* d_out, int out_size, void* d_ws, size_t ws_size,
                              hipStream_t stream) {
  const float* x  = (const float*)d_in[0];  // node_feature [4096,32,128]
  const float* WK = (const float*)d_in[1];  // [128,256]
  const float* WV = (const float*)d_in[2];  // [128,256]
  const float* WQ = (const float*)d_in[3];  // [128,256]
  float* outp = (float*)d_out;                              // [131072,128]
  float* wout = outp + (size_t)NGRAPH * NNODE * EMB;        // [131072,32,2]
  u32* ws = (u32*)d_ws;                                     // needs 192KB

  // 192 weight tiles, 8 waves/block -> 24 blocks
  prep_weights<<<24, 256, 0, stream>>>(WQ, WK, WV, ws);
  attn_sparsemax_kernel<<<NGRAPH, 256, SMEM_BYTES, stream>>>(x, ws, outp, wout);
}
